// _PositionAwareReductionKernel_38749194944561
// MI455X (gfx1250) — matrix operand order repaired and checked by reading, not compiled, not run on hardware
//
#include <hip/hip_runtime.h>
#include <hip/hip_bf16.h>
#include <math.h>

typedef _Float16 v16h __attribute__((ext_vector_type(16)));
typedef _Float16 v8h  __attribute__((ext_vector_type(8)));
typedef float    v8f  __attribute__((ext_vector_type(8)));
typedef float    v4f  __attribute__((ext_vector_type(4)));

#define BN      4096   // B*N rows
#define DIM     256
#define NOUT    512    // H*DH
#define NH      8
#define DH      64
#define SEQ     2048
#define NBH     16     // B*H

// ---------------------------------------------------------------------------
// Kernel 1: projection GEMM  proj = u @ W + bias   (M=4096, K=256, N=512)
// 256 threads = 8 waves; wave w computes a 16x16 tile at
// (row0 = by*128 + w*16, col0 = bx*16).  blockIdx.z selects q vs k.
// A (u) fragments loaded as two contiguous 8-float chunks per lane, cvt->f16.
// B (W) fragments loaded strided down a column (W is tiny, L2-resident).
// ---------------------------------------------------------------------------
__global__ void __launch_bounds__(256)
parope_proj_wmma(const float* __restrict__ u,
                 const float* __restrict__ Wq, const float* __restrict__ bq,
                 const float* __restrict__ Wk, const float* __restrict__ bk,
                 float* __restrict__ proj_q, float* __restrict__ proj_k) {
    const int lane   = threadIdx.x & 31;
    const int w      = threadIdx.x >> 5;
    const int lane15 = lane & 15;
    const int hi     = lane >> 4;

    const float* W    = blockIdx.z ? Wk : Wq;
    const float* bias = blockIdx.z ? bk : bq;
    float*       out  = blockIdx.z ? proj_k : proj_q;

    const int row0 = blockIdx.y * 128 + w * 16;
    const int col  = blockIdx.x * 16 + lane15;   // B/C fragment column
    const int arow = row0 + lane15;              // A fragment row

    const float* urow = u + (size_t)arow * DIM;

    v8f acc = {};
    #pragma unroll
    for (int kb = 0; kb < DIM; kb += 32) {
        // ---- A fragment: element i -> K = kb + 16*(i/8) + 8*hi + (i%8)
        v4f u0 = *(const v4f*)(urow + kb + hi * 8);
        v4f u1 = *(const v4f*)(urow + kb + hi * 8 + 4);
        v4f u2 = *(const v4f*)(urow + kb + 16 + hi * 8);
        v4f u3 = *(const v4f*)(urow + kb + 16 + hi * 8 + 4);
        v16h a;
        #pragma unroll
        for (int e = 0; e < 4; ++e) {
            a[e]      = (_Float16)u0[e];
            a[e + 4]  = (_Float16)u1[e];
            a[e + 8]  = (_Float16)u2[e];
            a[e + 12] = (_Float16)u3[e];
        }
        // ---- B fragment: element i -> K = kb + 16*(i/8) + 8*hi + (i%8), column = col
        v16h b;
        #pragma unroll
        for (int i = 0; i < 16; ++i)
            b[i] = (_Float16)W[(size_t)(kb + 16 * (i / 8) + hi * 8 + (i % 8)) * NOUT + col];

        acc = __builtin_amdgcn_wmma_f32_16x16x32_f16(
            false, a, false, b, (short)0, acc, false, false);
    }

    const float bv = bias[col];
    #pragma unroll
    for (int r = 0; r < 8; ++r)
        out[(size_t)(row0 + hi * 8 + r) * NOUT + col] = acc[r] + bv;
}

// ---------------------------------------------------------------------------
// Kernel 2: RoPE + transpose [b,n,h,d] -> [b,h,n,d] + narrow f32 -> f16.
// ---------------------------------------------------------------------------
__global__ void __launch_bounds__(256)
parope_rope(const float* __restrict__ proj_q, const float* __restrict__ proj_k,
            const float* __restrict__ pos,
            _Float16* __restrict__ qh, _Float16* __restrict__ kh) {
    int tid = blockIdx.x * blockDim.x + threadIdx.x;   // 2*B*N*H*DH threads
    int s = tid;
    const int d = s & 63;  s >>= 6;
    const int h = s & 7;   s >>= 3;
    const int n = s & 2047; s >>= 11;
    const int b = s & 1;   s >>= 1;
    const int qk = s;      // 0 = q, 1 = k

    const float* proj = qk ? proj_k : proj_q;
    _Float16*    out  = qk ? kh : qh;

    const size_t in_row = ((size_t)(b * SEQ + n)) * NOUT + h * DH;
    const float x = proj[in_row + d];
    const int   d2 = (d < 32) ? d + 32 : d - 32;
    const float rh = (d < 32) ? -proj[in_row + d2] : proj[in_row + d2];

    // inv_freq = 10000^(-(d%32)/32) ; t = pos[n] * (ROPE_SCALE/MIN_FREQ) = pos*64
    const float invf = exp2f(-(float)(d & 31) * 0.41524101186092029f); // log2(1e4)/32
    const float f = pos[n] * 64.0f * invf;
    float sf, cf;
    __sincosf(f, &sf, &cf);

    out[(((size_t)(b * NH + h) * SEQ) + n) * DH + d] = (_Float16)(x * cf + rh * sf);
}

// ---------------------------------------------------------------------------
// Kernel 3: scores[bh][i][j] = sum_d qh[bh][i][d] * kh[bh][j][d]
// 256 threads = 8 waves (2 along M x 4 along N); wave tile 32x32 (2x2 WMMA),
// block tile 64x128. K=64 -> two 16x16x32 steps. Fragments loaded directly
// from global (q/k are 8 MB total -> L2 resident). NT stores for the 256 MB
// streamed output.
// ---------------------------------------------------------------------------
__global__ void __launch_bounds__(256)
parope_qkt_wmma(const _Float16* __restrict__ qh, const _Float16* __restrict__ kh,
                float* __restrict__ out) {
    const int lane   = threadIdx.x & 31;
    const int w      = threadIdx.x >> 5;
    const int lane15 = lane & 15;
    const int hi     = lane >> 4;
    const int bh     = blockIdx.z;

    const int i0 = blockIdx.y * 64  + (w & 1) * 32;
    const int j0 = blockIdx.x * 128 + (w >> 1) * 32;

    const _Float16* qbase = qh + (size_t)bh * SEQ * DH;
    const _Float16* kbase = kh + (size_t)bh * SEQ * DH;

    v8f acc[2][2] = {{{}, {}}, {{}, {}}};

    #pragma unroll
    for (int ks = 0; ks < DH; ks += 32) {
        v16h a[2], b[2];
        #pragma unroll
        for (int mt = 0; mt < 2; ++mt) {
            const _Float16* qr = qbase + (size_t)(i0 + mt * 16 + lane15) * DH;
            a[mt] = *(const v16h*)(qr + ks + hi * 16);
        }
        #pragma unroll
        for (int nt = 0; nt < 2; ++nt) {
            const _Float16* kr = kbase + (size_t)(j0 + nt * 16 + lane15) * DH;
            b[nt] = *(const v16h*)(kr + ks + hi * 16);
        }
        #pragma unroll
        for (int mt = 0; mt < 2; ++mt)
            #pragma unroll
            for (int nt = 0; nt < 2; ++nt)
                acc[mt][nt] = __builtin_amdgcn_wmma_f32_16x16x32_f16(
                    false, a[mt], false, b[nt], (short)0, acc[mt][nt], false, false);
    }

    float* obase = out + (size_t)bh * SEQ * SEQ;
    #pragma unroll
    for (int mt = 0; mt < 2; ++mt) {
        #pragma unroll
        for (int nt = 0; nt < 2; ++nt) {
            const int col = j0 + nt * 16 + lane15;
            #pragma unroll
            for (int r = 0; r < 8; ++r) {
                const int row = i0 + mt * 16 + hi * 8 + r;
                __builtin_nontemporal_store(acc[mt][nt][r],
                                            obase + (size_t)row * SEQ + col);
            }
        }
    }
}

// ---------------------------------------------------------------------------
extern "C" void kernel_launch(void* const* d_in, const int* in_sizes, int n_in,
                              void* d_out, int out_size, void* d_ws, size_t ws_size,
                              hipStream_t stream) {
    const float* u   = (const float*)d_in[0];
    const float* pos = (const float*)d_in[1];
    const float* Wq  = (const float*)d_in[2];
    const float* bq  = (const float*)d_in[3];
    const float* Wk  = (const float*)d_in[4];
    const float* bk  = (const float*)d_in[5];
    float* out = (float*)d_out;

    char* ws = (char*)d_ws;
    float*    proj_q = (float*)(ws);                               // 8 MB
    float*    proj_k = (float*)(ws + (size_t)BN * NOUT * 4);       // 8 MB
    _Float16* qh     = (_Float16*)(ws + (size_t)BN * NOUT * 8);    // 4 MB
    _Float16* kh     = (_Float16*)(ws + (size_t)BN * NOUT * 8 + (size_t)NBH * SEQ * DH * 2);

    // 1) projections
    dim3 g1(NOUT / 16, BN / 128, 2);
    parope_proj_wmma<<<g1, 256, 0, stream>>>(u, Wq, bq, Wk, bk, proj_q, proj_k);

    // 2) RoPE + transpose + f16 narrow
    const int rope_threads = 2 * BN * NOUT; // 4,194,304
    parope_rope<<<rope_threads / 256, 256, 0, stream>>>(proj_q, proj_k, pos, qh, kh);

    // 3) Q @ K^T
    dim3 g3(SEQ / 128, SEQ / 64, NBH);
    parope_qkt_wmma<<<g3, 256, 0, stream>>>(qh, kh, out);
}
